// GCNModel_46127948759115
// MI455X (gfx1250) — compile-verified, matrix operand order checked by reading, not run on hardware
//
#include <hip/hip_runtime.h>

typedef __attribute__((ext_vector_type(16))) __bf16 v16bf;
typedef __attribute__((ext_vector_type(8)))  __bf16 v8bf;
typedef __attribute__((ext_vector_type(8)))  float  v8f;

#define B_TOT 8192
#define S_DIM 256
#define F_N   25
#define H_DIM 128
#define G_DIM 64
#define K_MLP 1600   // F_N * G_DIM

// workspace layout (bytes), all 16B aligned
#define OFF_W1T   0           // 128*256 bf16  = 65536 B
#define OFF_W2T   65536       //  64*128 bf16  = 16384 B
#define OFF_WM1T  81920       //  64*1600 bf16 = 204800 B
#define OFF_H2    286720      // 204800*64 bf16 = 26214400 B

// Fragment load: halves {0..7} and {16..23} relative to base (kb0 pre-folded).
static __device__ __forceinline__ v16bf ld2(const __bf16* base) {
  union { v16bf v; v8bf h[2]; } u;
  u.h[0] = *reinterpret_cast<const v8bf*>(base);
  u.h[1] = *reinterpret_cast<const v8bf*>(base + 16);
  return u.v;
}

static __device__ __forceinline__ v8f zero8() {
  v8f z;
#pragma unroll
  for (int i = 0; i < 8; ++i) z[i] = 0.f;
  return z;
}

static __device__ __forceinline__ v8f wmma_bf16(v16bf a, v16bf b, v8f c) {
  return __builtin_amdgcn_wmma_f32_16x16x32_bf16(false, a, false, b, (short)0,
                                                 c, false, false);
}

// ---------------------------------------------------------------------------
// Convert & transpose weights to bf16 [N][K] so B-fragments are contiguous.
__global__ void prep_weights(const float* __restrict__ W1,
                             const float* __restrict__ W2,
                             const float* __restrict__ Wm1,
                             __bf16* __restrict__ w1t,
                             __bf16* __restrict__ w2t,
                             __bf16* __restrict__ wm1t) {
  int i = blockIdx.x * blockDim.x + threadIdx.x;
  if (i < H_DIM * S_DIM) {                  // W1 [256,128] -> w1t [128][256]
    int n = i >> 8, k = i & 255;
    w1t[i] = (__bf16)W1[k * H_DIM + n];
  }
  if (i < G_DIM * H_DIM) {                  // W2 [128,64] -> w2t [64][128]
    int n = i >> 7, k = i & 127;
    w2t[i] = (__bf16)W2[k * G_DIM + n];
  }
  if (i < G_DIM * K_MLP) {                  // Wm1 [1600,64] -> wm1t [64][1600]
    int n = i / K_MLP, k = i - n * K_MLP;
    wm1t[i] = (__bf16)Wm1[k * 64 + n];
  }
}

// ---------------------------------------------------------------------------
// Fused: GEMM1 + bias + (ring-mix for batch 0) + ReLU + GEMM2 + bias + mix,
// emits H2 as bf16 [B*25][64]. One block = 2 batches (50 rows, padded to 64).
__launch_bounds__(256)
__global__ void gcn_fused(const float*  __restrict__ x,
                          const __bf16* __restrict__ w1t,
                          const float*  __restrict__ b1,
                          const __bf16* __restrict__ w2t,
                          const float*  __restrict__ b2,
                          __bf16*       __restrict__ h2out) {
  __shared__ __align__(16) __bf16 Xl[64 * 264];   // 33792 B (reused as Tl/Mixl)
  __shared__ __align__(16) __bf16 H1l[64 * 136];  // 17408 B
  float*  Tl   = reinterpret_cast<float*>(Xl);    // [64][68] f32
  __bf16* Mixl = Xl;                              // [25][128] bf16 temp

  const int t    = threadIdx.x;
  const int lane = t & 31;
  const int wave = t >> 5;
  const int ln   = lane & 15;
  const int kb0  = (lane >> 4) << 3;              // 0 or 8
  const int b0   = blockIdx.x * 2;

  // ---- Stage X: contiguous global reads, transpose (s,f)->(row=f, s) in LDS
  for (int idx = t; idx < 2 * S_DIM * F_N; idx += 256) {
    int bo  = idx / (S_DIM * F_N);
    int rem = idx - bo * (S_DIM * F_N);
    int s = rem / F_N, f = rem - s * F_N;
    Xl[(bo * F_N + f) * 264 + s] = (__bf16)x[(size_t)b0 * S_DIM * F_N + idx];
  }
  __syncthreads();

  // ---- GEMM1: [64x256] @ [256x128]; wave -> 2 M-tiles x 2 N-tiles
  {
    const int mtb = (wave & 1) * 2;       // 0 or 2
    const int ntb = (wave >> 1) * 2;      // 0,2,4,6
    const __bf16* abase[2];
    const __bf16* bbase[2];
#pragma unroll
    for (int mi = 0; mi < 2; ++mi)
      abase[mi] = &Xl[((mtb + mi) * 16 + ln) * 264] + kb0;
#pragma unroll
    for (int j = 0; j < 2; ++j)
      bbase[j] = w1t + ((ntb + j) * 16 + ln) * S_DIM + kb0;

    v8f acc[2][2];
#pragma unroll
    for (int mi = 0; mi < 2; ++mi)
#pragma unroll
      for (int j = 0; j < 2; ++j) acc[mi][j] = zero8();

    // software pipeline: prefetch k+1 while issuing k
    v16bf a[2], b[2];
#pragma unroll
    for (int mi = 0; mi < 2; ++mi) a[mi] = ld2(abase[mi]);
#pragma unroll
    for (int j = 0; j < 2; ++j) b[j] = ld2(bbase[j]);
#pragma unroll
    for (int ks = 0; ks < 8; ++ks) {
      v16bf an[2], bn[2];
      if (ks < 7) {
#pragma unroll
        for (int mi = 0; mi < 2; ++mi) an[mi] = ld2(abase[mi] + (ks + 1) * 32);
#pragma unroll
        for (int j = 0; j < 2; ++j) bn[j] = ld2(bbase[j] + (ks + 1) * 32);
      }
#pragma unroll
      for (int mi = 0; mi < 2; ++mi)
#pragma unroll
        for (int j = 0; j < 2; ++j)
          acc[mi][j] = wmma_bf16(a[mi], b[j], acc[mi][j]);
      if (ks < 7) {
#pragma unroll
        for (int mi = 0; mi < 2; ++mi) a[mi] = an[mi];
#pragma unroll
        for (int j = 0; j < 2; ++j) b[j] = bn[j];
      }
    }

#pragma unroll
    for (int mi = 0; mi < 2; ++mi) {
      const int rb = (mtb + mi) * 16 + (lane >> 4) * 8;
#pragma unroll
      for (int j = 0; j < 2; ++j) {
        int col = (ntb + j) * 16 + ln;
        float bias = b1[col];
#pragma unroll
        for (int i = 0; i < 8; ++i)
          H1l[(rb + i) * 136 + col] = (__bf16)(acc[mi][j][i] + bias); // pre-ReLU
      }
    }
  }
  __syncthreads();

  // ---- Ring aggregation for batch 0 (lives entirely in block 0), then ReLU
  if (blockIdx.x == 0) {
    for (int idx = t; idx < F_N * H_DIM; idx += 256) {
      int r = idx >> 7, c = idx & 127;
      int rm = (r + F_N - 1) % F_N, rp = (r + 1) % F_N;
      float v = ((float)H1l[rm * 136 + c] + (float)H1l[r * 136 + c] +
                 (float)H1l[rp * 136 + c]) * (1.f / 3.f);
      Mixl[idx] = (__bf16)v;
    }
    __syncthreads();
    for (int idx = t; idx < F_N * H_DIM; idx += 256) {
      int r = idx >> 7, c = idx & 127;
      float v = (float)Mixl[idx];
      H1l[r * 136 + c] = (__bf16)(v > 0.f ? v : 0.f);
    }
  }
  {
    int start = ((blockIdx.x == 0) ? F_N : 0) * H_DIM;
    for (int idx = t + start; idx < 50 * H_DIM; idx += 256) {
      int r = idx >> 7, c = idx & 127;
      float v = (float)H1l[r * 136 + c];
      H1l[r * 136 + c] = (__bf16)(v > 0.f ? v : 0.f);
    }
  }
  __syncthreads();

  // ---- GEMM2: [64x128] @ [128x64]; wave -> 2 M-tiles x 1 N-tile
  {
    const int mtb = (wave & 1) * 2;       // 0 or 2
    const int nt  = wave >> 1;            // 0..3
    const __bf16* abase[2];
#pragma unroll
    for (int mi = 0; mi < 2; ++mi)
      abase[mi] = &H1l[((mtb + mi) * 16 + ln) * 136] + kb0;
    const __bf16* bbase = w2t + (nt * 16 + ln) * H_DIM + kb0;

    v8f acc[2];
#pragma unroll
    for (int mi = 0; mi < 2; ++mi) acc[mi] = zero8();

    v16bf a[2], b;
#pragma unroll
    for (int mi = 0; mi < 2; ++mi) a[mi] = ld2(abase[mi]);
    b = ld2(bbase);
#pragma unroll
    for (int ks = 0; ks < 4; ++ks) {
      v16bf an[2], bn;
      if (ks < 3) {
#pragma unroll
        for (int mi = 0; mi < 2; ++mi) an[mi] = ld2(abase[mi] + (ks + 1) * 32);
        bn = ld2(bbase + (ks + 1) * 32);
      }
#pragma unroll
      for (int mi = 0; mi < 2; ++mi) acc[mi] = wmma_bf16(a[mi], b, acc[mi]);
      if (ks < 3) {
#pragma unroll
        for (int mi = 0; mi < 2; ++mi) a[mi] = an[mi];
        b = bn;
      }
    }

#pragma unroll
    for (int mi = 0; mi < 2; ++mi) {
      const int rb = (mtb + mi) * 16 + (lane >> 4) * 8;
      const int col = nt * 16 + ln;
#pragma unroll
      for (int i = 0; i < 8; ++i)
        Tl[(rb + i) * 68 + col] = acc[mi][i];                     // raw, no bias
    }
  }
  __syncthreads();

  // ---- Epilogue: mix (batch 0), + b2, emit bf16 H2 rows
  for (int idx = t; idx < 50 * G_DIM; idx += 256) {
    int r = idx >> 6, g = idx & 63;
    int bo = r / F_N;
    int f  = r - bo * F_N;
    int batch = b0 + bo;
    float v;
    if (batch == 0) {
      int rm = (f + F_N - 1) % F_N, rp = (f + 1) % F_N;
      v = (Tl[rm * 68 + g] + Tl[f * 68 + g] + Tl[rp * 68 + g]) * (1.f / 3.f);
    } else {
      v = Tl[r * 68 + g];
    }
    v += b2[g];
    h2out[((size_t)batch * F_N + f) * G_DIM + g] = (__bf16)v;
  }
}

// ---------------------------------------------------------------------------
// MLP head: [64 rows x 1600] @ [1600 x 64] (WMMA), ReLU, dot with Wm2 -> out
__launch_bounds__(256)
__global__ void mlp_head(const __bf16* __restrict__ h2,    // [8192][1600]
                         const __bf16* __restrict__ wm1t,  // [64][1600]
                         const float*  __restrict__ bm1,
                         const float*  __restrict__ wm2,   // [64]
                         const float*  __restrict__ bm2,
                         float*        __restrict__ out) {
  __shared__ __align__(16) float M1l[64 * 68];
  const int t    = threadIdx.x;
  const int lane = t & 31;
  const int wave = t >> 5;
  const int ln   = lane & 15;
  const int kb0  = (lane >> 4) << 3;
  const int rowb = blockIdx.x * 64;

  const int mt  = wave & 3;
  const int ntb = (wave >> 2) * 2;
  v8f acc[2];
#pragma unroll
  for (int j = 0; j < 2; ++j) acc[j] = zero8();

  const __bf16* ap  = h2 + (size_t)(rowb + mt * 16 + ln) * K_MLP + kb0;
  const __bf16* bp0 = wm1t + ((ntb + 0) * 16 + ln) * K_MLP + kb0;
  const __bf16* bp1 = wm1t + ((ntb + 1) * 16 + ln) * K_MLP + kb0;

  v16bf a   = ld2(ap);
  v16bf bb0 = ld2(bp0);
  v16bf bb1 = ld2(bp1);
  for (int ks = 0; ks < 49; ++ks) {
    ap += 32; bp0 += 32; bp1 += 32;
    v16bf an  = ld2(ap);
    v16bf bn0 = ld2(bp0);
    v16bf bn1 = ld2(bp1);
    acc[0] = wmma_bf16(a, bb0, acc[0]);
    acc[1] = wmma_bf16(a, bb1, acc[1]);
    a = an; bb0 = bn0; bb1 = bn1;
  }
  acc[0] = wmma_bf16(a, bb0, acc[0]);
  acc[1] = wmma_bf16(a, bb1, acc[1]);

  const int rb = mt * 16 + (lane >> 4) * 8;
#pragma unroll
  for (int j = 0; j < 2; ++j) {
    int col = (ntb + j) * 16 + ln;
    float bias = bm1[col];
#pragma unroll
    for (int i = 0; i < 8; ++i) {
      float v = acc[j][i] + bias;
      M1l[(rb + i) * 68 + col] = v > 0.f ? v : 0.f;
    }
  }
  __syncthreads();
  if (t < 64) {
    float s = 0.f;
#pragma unroll 8
    for (int c = 0; c < 64; ++c) s += M1l[t * 68 + c] * wm2[c];
    out[rowb + t] = s + bm2[0];
  }
}

// ---------------------------------------------------------------------------
extern "C" void kernel_launch(void* const* d_in, const int* in_sizes, int n_in,
                              void* d_out, int out_size, void* d_ws, size_t ws_size,
                              hipStream_t stream) {
  const float* x   = (const float*)d_in[0];
  const float* W1  = (const float*)d_in[1];
  const float* b1  = (const float*)d_in[2];
  const float* W2  = (const float*)d_in[3];
  const float* b2  = (const float*)d_in[4];
  const float* Wm1 = (const float*)d_in[5];
  const float* bm1 = (const float*)d_in[6];
  const float* Wm2 = (const float*)d_in[7];
  const float* bm2 = (const float*)d_in[8];

  char* ws = (char*)d_ws;
  __bf16* w1t  = (__bf16*)(ws + OFF_W1T);
  __bf16* w2t  = (__bf16*)(ws + OFF_W2T);
  __bf16* wm1t = (__bf16*)(ws + OFF_WM1T);
  __bf16* h2   = (__bf16*)(ws + OFF_H2);

  prep_weights<<<dim3(400), dim3(256), 0, stream>>>(W1, W2, Wm1, w1t, w2t, wm1t);
  gcn_fused<<<dim3(B_TOT / 2), dim3(256), 0, stream>>>(x, w1t, b1, w2t, b2, h2);
  mlp_head<<<dim3(B_TOT / 64), dim3(256), 0, stream>>>(h2, wm1t, bm1, Wm2, bm2,
                                                       (float*)d_out);
}
